// NeuralRenderer_18339510354769
// MI455X (gfx1250) — matrix operand order repaired and checked by reading, not compiled, not run on hardware
//
#include <hip/hip_runtime.h>
#include <hip/hip_bf16.h>

typedef __attribute__((ext_vector_type(16))) _Float16 v16h;
typedef __attribute__((ext_vector_type(8)))  float    v8f;

#define B_RAYS 8192
#define KC 64
#define HID 128

// ---------------------------------------------------------------------------
// z_coarse: z_steps = i/64 + noise/64 ; z = near*(1-t)+far*t
// ---------------------------------------------------------------------------
__global__ void zc_kernel(const float* __restrict__ rays,
                          const float* __restrict__ znoise,
                          float* __restrict__ z_coarse) {
  int idx = blockIdx.x * blockDim.x + threadIdx.x;   // B*KC threads
  int r = idx >> 6;
  int s = idx & 63;
  float nearv = rays[r * 8 + 6];
  float farv  = rays[r * 8 + 7];
  const float step = 1.0f / (float)KC;
  float t = (float)s * step + znoise[idx] * step;
  z_coarse[idx] = nearv * (1.0f - t) + farv * t;
}

// ---------------------------------------------------------------------------
// MLP: x[6] -> relu(x@W1+b1) -> relu(h@W2+b2) -> first 4 cols of W3 (+sigmoid)
// One workgroup = 128 threads = 4 waves; each wave owns 16 points (M=16).
// Layer2 = 8 Ntiles x 4 Kchunks of v_wmma_f32_16x16x32_f16.
// ---------------------------------------------------------------------------
__global__ __launch_bounds__(128)
void mlp_kernel(const float* __restrict__ rays,
                const float* __restrict__ zbuf, int n_samp,
                const float* __restrict__ w1, const float* __restrict__ b1,
                const float* __restrict__ w2, const float* __restrict__ b2,
                const float* __restrict__ w3, const float* __restrict__ b3,
                float* __restrict__ out4) {
  // W2 pre-swizzled into exact B-fragment layout: [ntile][kchunk][lane][16 halves]
  __shared__ __align__(32) _Float16 w2f[8][4][32][16];   // 32 KB
  __shared__ float w1s[6][HID];                          // 3 KB
  __shared__ float b1s[HID];
  __shared__ float b2s[HID];
  __shared__ float w3s[HID][4];                          // only 4 used outputs
  __shared__ float b3s[4];
  __shared__ float h2[4][16][HID];                       // 32 KB (per-wave slabs)

  const int tid = threadIdx.x;

  // ---- stage weights ----
  for (int i = tid; i < 8 * 4 * 32 * 16; i += 128) {
    int h  = i & 15;
    int l  = (i >> 4) & 31;
    int kc = (i >> 9) & 3;
    int n  = i >> 11;
    int K = kc * 32 + ((h < 8) ? h : h + 8) + ((l < 16) ? 0 : 8);  // B-frag: lanes<16 K={0..7,16..23}
    int N = n * 16 + (l & 15);
    w2f[n][kc][l][h] = (_Float16)w2[K * HID + N];
  }
  for (int i = tid; i < 6 * HID; i += 128) w1s[i / HID][i % HID] = w1[i];
  if (tid < HID) { b1s[tid] = b1[tid]; b2s[tid] = b2[tid]; }
  for (int i = tid; i < HID * 4; i += 128) w3s[i >> 2][i & 3] = w3[(i >> 2) * 36 + (i & 3)];
  if (tid < 4) b3s[tid] = b3[tid];
  __syncthreads();

  const int lane = tid & 31;
  const int wv   = tid >> 5;
  const int m    = lane & 15;

  // ---- per-point input (lanes l and l+16 duplicate the same row M) ----
  int point = blockIdx.x * 64 + wv * 16 + m;
  int ray   = point / n_samp;
  float o0 = rays[ray * 8 + 0], o1 = rays[ray * 8 + 1], o2 = rays[ray * 8 + 2];
  float d0 = rays[ray * 8 + 3], d1 = rays[ray * 8 + 4], d2 = rays[ray * 8 + 5];
  float z  = zbuf[point];
  float x[6] = { o0 + z * d0, o1 + z * d1, o2 + z * d2, d0, d1, d2 };

  // ---- layer 1 on VALU, written directly in A-fragment layout ----
  // 16-bit A 16x32: lanes<16 -> K {0..7,16..23}; lanes>=16 -> K {8..15,24..31}
  const int kln = (lane < 16) ? 0 : 8;
  v16h afrag[4];
  #pragma unroll
  for (int kc = 0; kc < 4; ++kc) {
    #pragma unroll
    for (int h = 0; h < 16; ++h) {
      int K = kc * 32 + ((h < 8) ? h : h + 8) + kln;
      float acc = b1s[K];
      #pragma unroll
      for (int j = 0; j < 6; ++j) acc += x[j] * w1s[j][K];
      afrag[kc][h] = (_Float16)fmaxf(acc, 0.0f);
    }
  }

  // ---- layer 2: WMMA 16x16x32, K = 128 in 4 chunks, N = 128 in 8 tiles ----
  const int ncol  = lane & 15;
  const int mrow0 = (lane < 16) ? 0 : 8;
  #pragma unroll
  for (int n = 0; n < 8; ++n) {
    v8f acc = {};
    #pragma unroll
    for (int kc = 0; kc < 4; ++kc) {
      v16h bfrag = *(const v16h*)&w2f[n][kc][lane][0];
      acc = __builtin_amdgcn_wmma_f32_16x16x32_f16(
          /*neg_a=*/false, afrag[kc], /*neg_b=*/false, bfrag,
          /*c_mod=*/(short)0, acc, /*reuse_a=*/false, /*reuse_b=*/false);
    }
    int N = n * 16 + ncol;
    float bias = b2s[N];
    #pragma unroll
    for (int r = 0; r < 8; ++r)
      h2[wv][mrow0 + r][N] = fmaxf(acc[r] + bias, 0.0f);  // C-layout: VGPR r -> M
  }
  __syncthreads();

  // ---- layer 3 (128x4) on VALU: lane -> (point pm, channel pair cp) ----
  const int pm = lane & 15;
  const int cp = lane >> 4;
  const int c0 = 2 * cp, c1 = 2 * cp + 1;
  float a0 = b3s[c0], a1 = b3s[c1];
  for (int K = 0; K < HID; ++K) {
    float hv = h2[wv][pm][K];
    a0 += hv * w3s[K][c0];
    a1 += hv * w3s[K][c1];
  }
  int p = blockIdx.x * 64 + wv * 16 + pm;
  float r0 = (c0 < 3) ? 1.0f / (1.0f + expf(-a0)) : a0;   // sigmoid rgb
  float r1 = (c1 < 3) ? 1.0f / (1.0f + expf(-a1)) : a1;   // c1==3 -> raw sigma
  out4[p * 4 + c0] = r0;
  out4[p * 4 + c1] = r1;
}

// ---------------------------------------------------------------------------
// Volume compositing: one thread per ray, serial transmittance scan.
// ---------------------------------------------------------------------------
__global__ void composite_kernel(const float* __restrict__ rays,
                                 const float* __restrict__ zbuf,
                                 const float* __restrict__ out4, int n_samp,
                                 float* __restrict__ w_out,
                                 float* __restrict__ rgb_out,
                                 float* __restrict__ depth_out) {
  int r = blockIdx.x * blockDim.x + threadIdx.x;
  if (r >= B_RAYS) return;
  float farv = rays[r * 8 + 7];
  float T = 1.0f, cr = 0.0f, cg = 0.0f, cb = 0.0f, dep = 0.0f;
  for (int s = 0; s < n_samp; ++s) {
    int pi = r * n_samp + s;
    float zs = zbuf[pi];
    float zn = (s < n_samp - 1) ? zbuf[pi + 1] : farv;
    float delta = zn - zs;
    float sg = fmaxf(out4[pi * 4 + 3], 0.0f);
    float alpha = 1.0f - expf(-delta * sg);
    float w = alpha * T;
    T *= (1.0f - alpha + 1e-10f);
    cr += w * out4[pi * 4 + 0];
    cg += w * out4[pi * 4 + 1];
    cb += w * out4[pi * 4 + 2];
    dep += w * zs;
    if (w_out) w_out[pi] = w;
  }
  rgb_out[r * 3 + 0] = cr;
  rgb_out[r * 3 + 1] = cg;
  rgb_out[r * 3 + 2] = cb;
  depth_out[r] = dep;
}

// ---------------------------------------------------------------------------
// Fine sampling: pdf/cdf, searchsorted(right), z_fine, stable rank-sort of 128.
// One workgroup (64 threads) per ray.
// ---------------------------------------------------------------------------
__global__ __launch_bounds__(64)
void sampler_kernel(const float* __restrict__ rays,
                    const float* __restrict__ z_coarse,
                    const float* __restrict__ w_c,
                    const float* __restrict__ u_fine,
                    const float* __restrict__ z_noise_f,
                    float* __restrict__ z_all) {
  __shared__ float scan[64];
  __shared__ float cdf[65];
  __shared__ float zl[128];
  const int r = blockIdx.x;
  const int t = threadIdx.x;

  scan[t] = w_c[r * 64 + t] + 1e-5f;
  __syncthreads();
  for (int off = 1; off < 64; off <<= 1) {       // Hillis-Steele inclusive scan
    float add = (t >= off) ? scan[t - off] : 0.0f;
    __syncthreads();
    scan[t] += add;
    __syncthreads();
  }
  float total = scan[63];
  if (t == 0) cdf[0] = 0.0f;
  cdf[t + 1] = scan[t] / total;
  __syncthreads();

  // searchsorted(cdf, u, side='right') over 65 entries
  float u = u_fine[r * 64 + t];
  int lo = 0, hi = 65;
  while (lo < hi) {
    int mid = (lo + hi) >> 1;
    if (cdf[mid] <= u) lo = mid + 1; else hi = mid;
  }
  float ind = fmaxf((float)lo - 1.0f, 0.0f);
  float nearv = rays[r * 8 + 6], farv = rays[r * 8 + 7];
  float st = (ind + z_noise_f[r * 64 + t]) * (1.0f / 64.0f);
  float zf = nearv * (1.0f - st) + farv * st;

  zl[t]      = z_coarse[r * 64 + t];
  zl[64 + t] = zf;
  __syncthreads();

  // stable rank-sort of 128 values -> z_all (sorted)
  #pragma unroll
  for (int e2 = 0; e2 < 2; ++e2) {
    int e = t + 64 * e2;
    float v = zl[e];
    int rank = 0;
    for (int j = 0; j < 128; ++j) {
      float o = zl[j];
      rank += (o < v) || (o == v && j < e);
    }
    z_all[r * 128 + rank] = v;
  }
}

// ---------------------------------------------------------------------------
extern "C" void kernel_launch(void* const* d_in, const int* in_sizes, int n_in,
                              void* d_out, int out_size, void* d_ws, size_t ws_size,
                              hipStream_t stream) {
  (void)in_sizes; (void)n_in; (void)out_size; (void)ws_size;
  const float* rays      = (const float*)d_in[0];
  const float* z_noise_c = (const float*)d_in[1];
  const float* u_fine    = (const float*)d_in[2];
  const float* z_noise_f = (const float*)d_in[3];
  const float* w1 = (const float*)d_in[4];
  const float* b1 = (const float*)d_in[5];
  const float* w2 = (const float*)d_in[6];
  const float* b2 = (const float*)d_in[7];
  const float* w3 = (const float*)d_in[8];
  const float* b3 = (const float*)d_in[9];
  float* out = (float*)d_out;

  float* ws       = (float*)d_ws;
  float* z_coarse = ws;                          // B*64
  float* w_c      = z_coarse + B_RAYS * 64;      // B*64
  float* z_all    = w_c + B_RAYS * 64;           // B*128
  float* pts_out  = z_all + B_RAYS * 128;        // B*128*4 (reused coarse/fine)

  float* rgb_c   = out;                          // 8192*3
  float* depth_c = rgb_c + B_RAYS * 3;           // 8192
  float* rgb_f   = depth_c + B_RAYS;             // 8192*3
  float* depth_f = rgb_f + B_RAYS * 3;           // 8192

  // coarse pass
  zc_kernel<<<(B_RAYS * 64) / 256, 256, 0, stream>>>(rays, z_noise_c, z_coarse);
  mlp_kernel<<<(B_RAYS * 64) / 64, 128, 0, stream>>>(rays, z_coarse, 64,
                                                     w1, b1, w2, b2, w3, b3, pts_out);
  composite_kernel<<<B_RAYS / 256, 256, 0, stream>>>(rays, z_coarse, pts_out, 64,
                                                     w_c, rgb_c, depth_c);
  // fine sampling + pass
  sampler_kernel<<<B_RAYS, 64, 0, stream>>>(rays, z_coarse, w_c, u_fine, z_noise_f, z_all);
  mlp_kernel<<<(B_RAYS * 128) / 64, 128, 0, stream>>>(rays, z_all, 128,
                                                      w1, b1, w2, b2, w3, b3, pts_out);
  composite_kernel<<<B_RAYS / 256, 256, 0, stream>>>(rays, z_all, pts_out, 128,
                                                     nullptr, rgb_f, depth_f);
}
